// EncoderLayer_82111184764994
// MI455X (gfx1250) — matrix operand order repaired and checked by reading, not compiled, not run on hardware
//
#include <hip/hip_runtime.h>
#include <hip/hip_bf16.h>
#include <math.h>

// ---------------- problem constants ----------------
#define B_   2
#define S_   2048
#define D_   1024
#define H_   16
#define HD_  64
#define PF_  4096
#define N_   (B_ * S_)        // 4096 rows of activations
#define DH2_ (D_ / 2)         // 512 rope freqs

typedef __attribute__((ext_vector_type(16))) __bf16 bf16x16;
typedef __attribute__((ext_vector_type(8)))  __bf16 bf16x8;
typedef __attribute__((ext_vector_type(8)))  float  f32x8;

// ---------------- WMMA helpers ----------------
__device__ __forceinline__ f32x8 wmma_bf16(bf16x16 a, bf16x16 b, f32x8 c) {
    return __builtin_amdgcn_wmma_f32_16x16x32_bf16(
        false, a, false, b, (short)0, c, false, false);
}

// A-matrix 16x32 bf16 fragment: lane l row M=l%16,
// element e: K = (l/16)*16 + e -> two contiguous 16B chunks.
__device__ __forceinline__ bf16x16 load_a_frag(const __bf16* p) {
    int half16 = (threadIdx.x & 31) >> 4;
    bf16x8 lo = *(const bf16x8*)(p + half16 * 16);
    bf16x8 hi = *(const bf16x8*)(p + 8 + half16 * 16);
    bf16x16 f;
#pragma unroll
    for (int i = 0; i < 8; ++i) { f[i] = lo[i]; f[i + 8] = hi[i]; }
    return f;
}

// async global -> LDS copy, 16B per lane (ASYNCcnt-tracked)
__device__ __forceinline__ void async_copy_b128(uint32_t lds_addr, const void* g) {
    asm volatile("global_load_async_to_lds_b128 %0, %1, off"
                 :: "v"(lds_addr), "v"((unsigned long long)(uintptr_t)g)
                 : "memory");
}

// ---------------- elementwise converters ----------------
__global__ void cvt_f32_bf16_kernel(const float* __restrict__ x,
                                    __bf16* __restrict__ y, size_t n) {
    size_t t = (size_t)blockIdx.x * blockDim.x + threadIdx.x;
    if (t < n) y[t] = (__bf16)x[t];
}

// Pack row-major W[Kd x Nd] (f32) -> WMMA B-fragment tiles (bf16).
__global__ void pack_w_kernel(const float* __restrict__ W,
                              __bf16* __restrict__ Wp, int Kd, int Nd) {
    int t    = blockIdx.x * blockDim.x + threadIdx.x;
    int lane = t & 31;
    int tile = t >> 5;
    int ntn  = Nd >> 4;
    int kt   = tile / ntn;
    int nt   = tile - kt * ntn;
    if (kt >= (Kd >> 5)) return;
    int n  = nt * 16 + (lane & 15);
    int kb = kt * 32 + ((lane >> 4) << 4);
    bf16x16 vv;
#pragma unroll
    for (int e = 0; e < 16; ++e)
        vv[e] = (__bf16)W[(size_t)(kb + e) * Nd + n];
    *(bf16x16*)(Wp + ((size_t)tile * 32 + lane) * 16) = vv;
}

// ---------------- generic WMMA GEMM ----------------
// Wave computes 32 rows x 64 cols; block = 8 waves = 256 rows x 64 cols.
// B tiles staged in LDS via double-buffered async copies.
// MODE 0: f32 out (+opt residual)  MODE 1: bf16 out  MODE 2: relu->bf16
template <int MODE>
__global__ __launch_bounds__(256) void gemm_wmma_kernel(
    const __bf16* __restrict__ A, const __bf16* __restrict__ Bp,
    const float* __restrict__ bias, const float* __restrict__ resid,
    float* __restrict__ outF, __bf16* __restrict__ outB,
    int M, int K, int Nd) {
    __shared__ alignas(128) __bf16 Bs[2][4 * 512];   // 2 bufs x (4 tiles x 1KB)
    int tid  = threadIdx.x;
    int lane = tid & 31;
    int wave = tid >> 5;
    int row0 = blockIdx.x * 256 + wave * 32;
    int col0 = blockIdx.y * 64;
    int ntn  = Nd >> 4;
    const __bf16* arow0 = A + (size_t)(row0 + (lane & 15)) * K;
    const __bf16* arow1 = arow0 + (size_t)16 * K;

    uint32_t ldsB0 = (uint32_t)(uintptr_t)&Bs[0][0];
    uint32_t ldsB1 = (uint32_t)(uintptr_t)&Bs[1][0];
    const __bf16* bsrc = Bp + (size_t)(col0 >> 4) * 512;   // + kt*ntn*512

    int KT = K >> 5;
    // prologue: prefetch kt=0,1
    async_copy_b128(ldsB0 + tid * 16,
                    (const char*)(bsrc) + tid * 16);
    if (KT > 1)
        async_copy_b128(ldsB1 + tid * 16,
                        (const char*)(bsrc + (size_t)ntn * 512) + tid * 16);

    f32x8 acc[2][4] = {};
    for (int kt = 0; kt < KT; ++kt) {
        int p = kt & 1;
        if (kt < KT - 1) asm volatile("s_wait_asynccnt 0x1" ::: "memory");
        else             asm volatile("s_wait_asynccnt 0x0" ::: "memory");
        __syncthreads();
        bf16x16 a0 = load_a_frag(arow0 + kt * 32);
        bf16x16 a1 = load_a_frag(arow1 + kt * 32);
#pragma unroll
        for (int nt = 0; nt < 4; ++nt) {
            bf16x16 b = *(const bf16x16*)(&Bs[p][(nt * 32 + lane) * 16]);
            acc[0][nt] = wmma_bf16(a0, b, acc[0][nt]);
            acc[1][nt] = wmma_bf16(a1, b, acc[1][nt]);
        }
        __syncthreads();
        if (kt + 2 < KT)
            async_copy_b128((p ? ldsB0 : ldsB1) + tid * 16,
                            (const char*)(bsrc + (size_t)(kt + 2) * ntn * 512) + tid * 16);
    }

    // C layout: lane -> col = lane%16, rows = (lane/16)*8 + v
#pragma unroll
    for (int hh = 0; hh < 2; ++hh) {
        int rowbase = row0 + hh * 16 + ((lane >> 4) << 3);
#pragma unroll
        for (int nt = 0; nt < 4; ++nt) {
            int c = col0 + nt * 16 + (lane & 15);
            float bv = bias ? bias[c] : 0.0f;
#pragma unroll
            for (int v = 0; v < 8; ++v) {
                size_t idx = (size_t)(rowbase + v) * Nd + c;
                float val  = acc[hh][nt][v] + bv;
                if (MODE == 0) {
                    if (resid) val += resid[idx];
                    outF[idx] = val;
                } else if (MODE == 1) {
                    outB[idx] = (__bf16)val;
                } else {
                    outB[idx] = (__bf16)fmaxf(val, 0.0f);
                }
            }
        }
    }
}

// ---------------- RoPE + head-major rearrange ----------------
__global__ void rope_rearrange_kernel(const __bf16* __restrict__ Qt,
                                      const __bf16* __restrict__ Kt,
                                      const float* __restrict__ cosT,
                                      const float* __restrict__ sinT,
                                      __bf16* __restrict__ Qh,
                                      __bf16* __restrict__ Kh) {
    int t = blockIdx.x * blockDim.x + threadIdx.x;
    if (t >= N_ * DH2_) return;
    int i  = t & (DH2_ - 1);
    int bs = t / DH2_;
    int s  = bs & (S_ - 1);
    int b  = bs / S_;
    float c  = cosT[(size_t)s * DH2_ + i];
    float sn = sinT[(size_t)s * DH2_ + i];
    size_t src0 = (size_t)bs * D_ + 2 * i;
    int col = 2 * i;
    int h   = col >> 6;
    int hd  = col & 63;
    size_t dst = ((size_t)(b * H_ + h) * S_ + s) * HD_ + hd;
    {
        float a = (float)Qt[src0], bb = (float)Qt[src0 + 1];
        Qh[dst]     = (__bf16)(a * c - bb * sn);
        Qh[dst + 1] = (__bf16)(a * sn + bb * c);
    }
    {
        float a = (float)Kt[src0], bb = (float)Kt[src0 + 1];
        Kh[dst]     = (__bf16)(a * c - bb * sn);
        Kh[dst + 1] = (__bf16)(a * sn + bb * c);
    }
}

// Pack V into per-(b,h) B-fragment tiles.
__global__ void pack_v_kernel(const __bf16* __restrict__ Vt,
                              __bf16* __restrict__ Vp) {
    int t    = blockIdx.x * blockDim.x + threadIdx.x;
    int lane = t & 31;
    int rest = t >> 5;
    int dt   = rest & 3;  rest >>= 2;
    int kt   = rest & ((S_ / 32) - 1);
    int bh   = rest / (S_ / 32);
    if (bh >= B_ * H_) return;
    int b = bh / H_, h = bh - b * H_;
    int srow = kt * 32 + ((lane >> 4) << 3);
    int col  = h * HD_ + dt * 16 + (lane & 15);
    bf16x16 vv;
#pragma unroll
    for (int e = 0; e < 16; ++e)
        vv[e] = Vt[(size_t)(b * S_ + srow + e + (e & 8)) * D_ + col];
    *(bf16x16*)(Vp + (((size_t)(bh * (S_ / 32) + kt) * 4 + dt) * 32 + lane) * 16) = vv;
}

// ---------------- flash attention, transposed-score formulation ----------------
// S^T = K * Q^T  ->  each lane owns one query column; softmax reductions are
// in-lane + one shfl_xor(16).  P^T's C-register layout IS the A-fragment layout
// for O += P*V, so no LDS transpose is needed.  Waves fully independent.
__global__ __launch_bounds__(128) void attn_kernel(
    const __bf16* __restrict__ Qh, const __bf16* __restrict__ Kh,
    const __bf16* __restrict__ Vp, __bf16* __restrict__ attnout) {
    int lane = threadIdx.x & 31;
    int wave = threadIdx.x >> 5;
    int bh   = blockIdx.x;
    int b    = bh / H_, h = bh - b * H_;
    int q0   = blockIdx.y * 64 + wave * 16;

    // B-fragment of Q^T: lane -> query col = q0 + lane%16, contiguous hd chunk
    const __bf16* qptr =
        Qh + ((size_t)bh * S_ + q0 + (lane & 15)) * HD_ + ((lane >> 4) << 4);
    bf16x16 bq0 = *(const bf16x16*)(qptr);        // hd 0..31 slice
    bf16x16 bq1 = *(const bf16x16*)(qptr + 32);   // hd 32..63 slice

    f32x8 O[4] = {};
    float mrun = -1e30f, ssum = 0.0f;     // per-lane: query q0 + lane%16

    const __bf16* kbaseP = Kh + (size_t)bh * S_ * HD_;
    const __bf16* vbaseP = Vp + (size_t)bh * (S_ / 32) * 2048;
    const float scale = 0.125f;           // 1/sqrt(64)
    const int rowq = (lane >> 4) << 3;    // first row-query this lane accumulates

    for (int kb = 0; kb < S_; kb += 32) {
        // ---- S^T tiles: keys kb..kb+15 (st0) and kb+16..kb+31 (st1) ----
        const __bf16* k0 = kbaseP + (size_t)(kb + (lane & 15)) * HD_;
        const __bf16* k1 = k0 + 16 * HD_;
        f32x8 st0 = {}, st1 = {};
        st0 = wmma_bf16(load_a_frag(k0),      bq0, st0);
        st0 = wmma_bf16(load_a_frag(k0 + 32), bq1, st0);
        st1 = wmma_bf16(load_a_frag(k1),      bq0, st1);
        st1 = wmma_bf16(load_a_frag(k1 + 32), bq1, st1);

        // ---- online softmax over this lane's query column ----
        float mx = -1e30f;
#pragma unroll
        for (int v = 0; v < 8; ++v) {
            st0[v] *= scale; st1[v] *= scale;
            mx = fmaxf(mx, fmaxf(st0[v], st1[v]));
        }
        mx = fmaxf(mx, __shfl_xor(mx, 16));
        float mn   = fmaxf(mrun, mx);
        float corr = __expf(mrun - mn);
        mrun = mn;

        bf16x16 ap;          // P as A-fragment, directly from C registers
        float rs = 0.0f;
#pragma unroll
        for (int v = 0; v < 8; ++v) {
            float p0 = __expf(st0[v] - mn);
            float p1 = __expf(st1[v] - mn);
            rs += p0 + p1;
            ap[v]     = (__bf16)p0;
            ap[v + 8] = (__bf16)p1;
        }
        rs += __shfl_xor(rs, 16);
        ssum = ssum * corr + rs;

        // broadcast corr from column space into this lane's 8 row-queries
#pragma unroll
        for (int v = 0; v < 8; ++v) {
            float cr = __shfl(corr, rowq + v);
            O[0][v] *= cr; O[1][v] *= cr; O[2][v] *= cr; O[3][v] *= cr;
        }

        // ---- O += P * V (packed V fragments, one 32B load each) ----
        const __bf16* vp = vbaseP + (size_t)(kb >> 5) * 2048 + lane * 16;
        O[0] = wmma_bf16(ap, *(const bf16x16*)(vp),        O[0]);
        O[1] = wmma_bf16(ap, *(const bf16x16*)(vp + 512),  O[1]);
        O[2] = wmma_bf16(ap, *(const bf16x16*)(vp + 1024), O[2]);
        O[3] = wmma_bf16(ap, *(const bf16x16*)(vp + 1536), O[3]);
    }

    // ---- finalize: divide by row sum, scatter to (b,s,D) bf16 ----
    float inv[8];
#pragma unroll
    for (int v = 0; v < 8; ++v)
        inv[v] = 1.0f / __shfl(ssum, rowq + v);
#pragma unroll
    for (int dt = 0; dt < 4; ++dt) {
        int col = h * HD_ + dt * 16 + (lane & 15);
#pragma unroll
        for (int v = 0; v < 8; ++v) {
            int q = q0 + rowq + v;
            attnout[(size_t)(b * S_ + q) * D_ + col] = (__bf16)(O[dt][v] * inv[v]);
        }
    }
}

// ---------------- row LayerNorm ----------------
__global__ __launch_bounds__(256) void ln_kernel(
    const float* __restrict__ x, const float* __restrict__ g,
    const float* __restrict__ bta, float* __restrict__ outF,
    __bf16* __restrict__ outB, int Dd) {
    __shared__ float red[2][8];
    int row = blockIdx.x;
    const float* xr = x + (size_t)row * Dd;
    float s = 0.0f, s2 = 0.0f;
    for (int i = threadIdx.x; i < Dd; i += blockDim.x) {
        float v = xr[i]; s += v; s2 += v * v;
    }
    for (int m = 1; m < 32; m <<= 1) {
        s += __shfl_xor(s, m); s2 += __shfl_xor(s2, m);
    }
    int wave = threadIdx.x >> 5, lane = threadIdx.x & 31;
    if (lane == 0) { red[0][wave] = s; red[1][wave] = s2; }
    __syncthreads();
    if (threadIdx.x == 0) {
        float a = 0.0f, c = 0.0f;
        for (int i = 0; i < 8; ++i) { a += red[0][i]; c += red[1][i]; }
        red[0][0] = a; red[1][0] = c;
    }
    __syncthreads();
    float mean = red[0][0] / Dd;
    float var  = red[1][0] / Dd - mean * mean;
    float inv  = rsqrtf(var + 1e-5f);
    for (int i = threadIdx.x; i < Dd; i += blockDim.x) {
        float v = (xr[i] - mean) * inv * g[i] + bta[i];
        if (outF) outF[(size_t)row * Dd + i] = v;
        if (outB) outB[(size_t)row * Dd + i] = (__bf16)v;
    }
}

// ---------------- host side ----------------
extern "C" void kernel_launch(void* const* d_in, const int* in_sizes, int n_in,
                              void* d_out, int out_size, void* d_ws, size_t ws_size,
                              hipStream_t stream) {
    (void)in_sizes; (void)n_in; (void)out_size; (void)ws_size;
    const float* src  = (const float*)d_in[0];
    const float* Wq   = (const float*)d_in[1];
    const float* bq   = (const float*)d_in[2];
    const float* Wk   = (const float*)d_in[3];
    const float* bk   = (const float*)d_in[4];
    const float* Wv   = (const float*)d_in[5];
    const float* bv   = (const float*)d_in[6];
    const float* Wo   = (const float*)d_in[7];
    const float* bo   = (const float*)d_in[8];
    const float* ln1g = (const float*)d_in[9];
    const float* ln1b = (const float*)d_in[10];
    const float* W1   = (const float*)d_in[11];
    const float* b1   = (const float*)d_in[12];
    const float* W2   = (const float*)d_in[13];
    const float* b2   = (const float*)d_in[14];
    const float* ln2g = (const float*)d_in[15];
    const float* ln2b = (const float*)d_in[16];
    const float* cosT = (const float*)d_in[17];
    const float* sinT = (const float*)d_in[18];

    char* w = (char*)d_ws;
    auto alloc = [&](size_t bytes) -> void* {
        void* p = (void*)w;
        w += (bytes + 255) & ~(size_t)255;
        return p;
    };
    const size_t ND = (size_t)N_ * D_;
    __bf16* Xbf  = (__bf16*)alloc(ND * 2);
    __bf16* Wqp  = (__bf16*)alloc((size_t)D_ * D_ * 2);
    __bf16* Wkp  = (__bf16*)alloc((size_t)D_ * D_ * 2);
    __bf16* Wvp  = (__bf16*)alloc((size_t)D_ * D_ * 2);
    __bf16* Wop  = (__bf16*)alloc((size_t)D_ * D_ * 2);
    __bf16* W1p  = (__bf16*)alloc((size_t)D_ * PF_ * 2);
    __bf16* W2p  = (__bf16*)alloc((size_t)PF_ * D_ * 2);
    __bf16* Qt   = (__bf16*)alloc(ND * 2);
    __bf16* Kt   = (__bf16*)alloc(ND * 2);
    __bf16* Vt   = (__bf16*)alloc(ND * 2);
    __bf16* Qh   = (__bf16*)alloc(ND * 2);
    __bf16* Kh   = (__bf16*)alloc(ND * 2);
    __bf16* Vp   = (__bf16*)alloc(ND * 2);
    __bf16* aout = (__bf16*)alloc(ND * 2);
    float*  y1   = (float*)alloc(ND * 4);
    float*  hf   = (float*)alloc(ND * 4);
    __bf16* hbf  = (__bf16*)alloc(ND * 2);
    __bf16* act  = (__bf16*)alloc((size_t)N_ * PF_ * 2);
    float*  y2   = (float*)alloc(ND * 4);

    cvt_f32_bf16_kernel<<<(unsigned)(ND / 256), 256, 0, stream>>>(src, Xbf, ND);

    pack_w_kernel<<<(D_/32)*(D_/16)*32/256, 256, 0, stream>>>(Wq, Wqp, D_, D_);
    pack_w_kernel<<<(D_/32)*(D_/16)*32/256, 256, 0, stream>>>(Wk, Wkp, D_, D_);
    pack_w_kernel<<<(D_/32)*(D_/16)*32/256, 256, 0, stream>>>(Wv, Wvp, D_, D_);
    pack_w_kernel<<<(D_/32)*(D_/16)*32/256, 256, 0, stream>>>(Wo, Wop, D_, D_);
    pack_w_kernel<<<(D_/32)*(PF_/16)*32/256, 256, 0, stream>>>(W1, W1p, D_, PF_);
    pack_w_kernel<<<(PF_/32)*(D_/16)*32/256, 256, 0, stream>>>(W2, W2p, PF_, D_);

    dim3 gQKV(N_ / 256, D_ / 64);
    gemm_wmma_kernel<1><<<gQKV, 256, 0, stream>>>(Xbf, Wqp, bq, nullptr, nullptr, Qt, N_, D_, D_);
    gemm_wmma_kernel<1><<<gQKV, 256, 0, stream>>>(Xbf, Wkp, bk, nullptr, nullptr, Kt, N_, D_, D_);
    gemm_wmma_kernel<1><<<gQKV, 256, 0, stream>>>(Xbf, Wvp, bv, nullptr, nullptr, Vt, N_, D_, D_);

    rope_rearrange_kernel<<<(N_ * DH2_) / 256, 256, 0, stream>>>(Qt, Kt, cosT, sinT, Qh, Kh);
    pack_v_kernel<<<(B_*H_*(S_/32)*4*32) / 256, 256, 0, stream>>>(Vt, Vp);

    attn_kernel<<<dim3(B_ * H_, S_ / 64), 128, 0, stream>>>(Qh, Kh, Vp, aout);

    gemm_wmma_kernel<0><<<gQKV, 256, 0, stream>>>(aout, Wop, bo, src, y1, nullptr, N_, D_, D_);
    ln_kernel<<<N_, 256, 0, stream>>>(y1, ln1g, ln1b, hf, hbf, D_);

    dim3 gF1(N_ / 256, PF_ / 64);
    gemm_wmma_kernel<2><<<gF1, 256, 0, stream>>>(hbf, W1p, b1, nullptr, nullptr, act, N_, D_, PF_);
    dim3 gF2(N_ / 256, D_ / 64);
    gemm_wmma_kernel<0><<<gF2, 256, 0, stream>>>(act, W2p, b2, hf, y2, nullptr, N_, PF_, D_);
    ln_kernel<<<N_, 256, 0, stream>>>(y2, ln2g, ln2b, (float*)d_out, nullptr, D_);
}
